// Encoder_13219909337091
// MI455X (gfx1250) — matrix operand order repaired and checked by reading, not compiled, not run on hardware
//
#include <hip/hip_runtime.h>
#include <math.h>

// ---------------------------------------------------------------------------
// Types
// ---------------------------------------------------------------------------
typedef __bf16 bf16;
typedef __attribute__((ext_vector_type(8)))  bf16  v8bf;
typedef __attribute__((ext_vector_type(16))) bf16  v16bf;
typedef __attribute__((ext_vector_type(8)))  float v8f;

#define BM 128
#define BN 128
#define BKK 32
#define LDSS 40   // 32 data bf16 + 8 pad (80B rows keep 16B alignment, kill bank conflicts)

// ---------------------------------------------------------------------------
// Core WMMA GEMM body:  out = act( A[M,K] * Bt[N,K]^T [+ bias[N]] [+ addend[M,N]] )
// A, Bt bf16 row-major, contiguous K. 256 threads, block tile 128x128,
// 8 waves -> each wave a 32(M) x 64(N) region = 2x4 wmma tiles.
// All configuration is compile-time -> branchless epilogue.
// ---------------------------------------------------------------------------
template <int ACT, bool HAS_BIAS, bool HAS_ADD, bool OUTF, bool OUTB>
__device__ __forceinline__ void gemm_body(
    const bf16* __restrict__ A, const bf16* __restrict__ Bt,
    const float* __restrict__ bias, const bf16* __restrict__ addend,
    float* __restrict__ outF, bf16* __restrict__ outB,
    int N, int K)
{
    __shared__ bf16 sA[BM * LDSS];
    __shared__ bf16 sB[BN * LDSS];

    const int tid   = threadIdx.x;
    const int lane  = tid & 31;
    const int wave  = tid >> 5;
    const int waveM = wave & 3;          // 4 waves along M
    const int waveN = wave >> 2;         // 2 waves along N
    const int fr    = lane & 15;
    const int fh    = lane >> 4;         // 0/1 (K-half selector per ISA layout)

    const int bm = blockIdx.y * BM;
    const int bn = blockIdx.x * BN;

    // cooperative tile load mapping: 256 threads x 32B = 128x32 bf16 tile
    const int ldrow = tid >> 1;          // 0..127
    const int ldseg = (tid & 1) * 16;    // 0 or 16 (bf16 elements)

    v8f acc[2][4];
#pragma unroll
    for (int tm = 0; tm < 2; ++tm)
#pragma unroll
        for (int tn = 0; tn < 4; ++tn)
#pragma unroll
            for (int i = 0; i < 8; ++i) acc[tm][tn][i] = 0.0f;

    const bf16* Ag = A  + (size_t)(bm + ldrow) * K + ldseg;
    const bf16* Bg = Bt + (size_t)(bn + ldrow) * K + ldseg;

    for (int k0 = 0; k0 < K; k0 += BKK) {
        // ---- global -> LDS (two b128 loads per tile per thread) ----
        v8bf a0 = *(const v8bf*)(Ag + k0);
        v8bf a1 = *(const v8bf*)(Ag + k0 + 8);
        v8bf b0 = *(const v8bf*)(Bg + k0);
        v8bf b1 = *(const v8bf*)(Bg + k0 + 8);
        if (k0 + BKK < K) {                     // gfx1250 global_prefetch_b8
            __builtin_prefetch(Ag + k0 + BKK, 0, 3);
            __builtin_prefetch(Bg + k0 + BKK, 0, 3);
        }
        *(v8bf*)&sA[ldrow * LDSS + ldseg]     = a0;
        *(v8bf*)&sA[ldrow * LDSS + ldseg + 8] = a1;
        *(v8bf*)&sB[ldrow * LDSS + ldseg]     = b0;
        *(v8bf*)&sB[ldrow * LDSS + ldseg + 8] = b1;
        __syncthreads();

        // ---- fragments (ISA 7.12.2 16-bit layouts) ----
        // A 16x32: lanes0-15 K={0..7,16..23}, lanes16-31 K={8..15,24..31}
        v16bf afrag[2];
#pragma unroll
        for (int tm = 0; tm < 2; ++tm) {
            const int r = waveM * 32 + tm * 16 + fr;
            v8bf lo = *(const v8bf*)&sA[r * LDSS + fh * 8];
            v8bf hi = *(const v8bf*)&sA[r * LDSS + 16 + fh * 8];
#pragma unroll
            for (int i = 0; i < 8; ++i) { afrag[tm][i] = lo[i]; afrag[tm][8 + i] = hi[i]; }
        }
        // B 32x16: lanes0-15 K={0..7,16..23}, lanes16-31 K={8..15,24..31}
        v16bf bfrag[4];
#pragma unroll
        for (int tn = 0; tn < 4; ++tn) {
            const int c = waveN * 64 + tn * 16 + fr;
            v8bf lo = *(const v8bf*)&sB[c * LDSS + fh * 8];
            v8bf hi = *(const v8bf*)&sB[c * LDSS + 16 + fh * 8];
#pragma unroll
            for (int i = 0; i < 8; ++i) { bfrag[tn][i] = lo[i]; bfrag[tn][8 + i] = hi[i]; }
        }

#pragma unroll
        for (int tn = 0; tn < 4; ++tn)
#pragma unroll
            for (int tm = 0; tm < 2; ++tm)
                acc[tm][tn] = __builtin_amdgcn_wmma_f32_16x16x32_bf16(
                    false, afrag[tm], false, bfrag[tn],
                    (short)0, acc[tm][tn], false, false);

        __syncthreads();
    }

    // ---- epilogue: branchless (compile-time config) ----
#pragma unroll
    for (int tm = 0; tm < 2; ++tm) {
#pragma unroll
        for (int tn = 0; tn < 4; ++tn) {
            const int row0 = bm + waveM * 32 + tm * 16 + fh * 8;
            const int col  = bn + waveN * 64 + tn * 16 + fr;
            float bv = 0.0f;
            if (HAS_BIAS) bv = bias[col];
#pragma unroll
            for (int i = 0; i < 8; ++i) {
                const size_t off = (size_t)(row0 + i) * N + col;
                float v = acc[tm][tn][i] + bv;
                if (HAS_ADD) v += (float)addend[off];
                if (ACT == 1)      v = 1.0f / (1.0f + __expf(-v));
                else if (ACT == 2) v = tanhf(v);
                if (OUTF) outF[off] = v;
                if (OUTB) outB[off] = (bf16)v;
            }
        }
    }
}

// ---------------------------------------------------------------------------
// Kernels
// ---------------------------------------------------------------------------
template <int ACT>
__global__ __launch_bounds__(256) void wmma_gemm_kernel(
    const bf16* __restrict__ A, const bf16* __restrict__ Bt,
    const float* __restrict__ bias, bf16* __restrict__ outB, int N, int K)
{
    gemm_body<ACT, true, false, false, true>(A, Bt, bias, nullptr, nullptr,
                                             outB, N, K);
}

// One launch computes all 4 recurrent gates (blockIdx.z selects gate).
// gates[g] = act_g( h @ Wg^T + Xg_t ) ; g=0:i,1:f,2:o (sigmoid), 3:ctilde (tanh)
__global__ __launch_bounds__(256) void recurrent_gemm_kernel(
    const bf16* __restrict__ Hbf,
    const bf16* __restrict__ W0, const bf16* __restrict__ W1,
    const bf16* __restrict__ W2, const bf16* __restrict__ W3,
    const bf16* __restrict__ X0, const bf16* __restrict__ X1,
    const bf16* __restrict__ X2, const bf16* __restrict__ X3,
    float* __restrict__ gates, int Bn, int Hn)
{
    const int g = blockIdx.z;
    const bf16* Bt = (g == 0) ? W0 : (g == 1) ? W1 : (g == 2) ? W2 : W3;
    const bf16* Xg = (g == 0) ? X0 : (g == 1) ? X1 : (g == 2) ? X2 : X3;
    float* outF = gates + (size_t)g * Bn * Hn;
    if (g == 3)
        gemm_body<2, false, true, true, false>(Hbf, Bt, nullptr, Xg, outF,
                                               nullptr, Hn, Hn);
    else
        gemm_body<1, false, true, true, false>(Hbf, Bt, nullptr, Xg, outF,
                                               nullptr, Hn, Hn);
}

// c = f*c + i*ctilde + l*zhat ; h = o*tanh(c)
__global__ void pointwise_step_kernel(
    const float* __restrict__ gates, const bf16* __restrict__ l_t,
    const bf16* __restrict__ zh_t, float* __restrict__ cbuf,
    float* __restrict__ hout, bf16* __restrict__ hbf, int n)
{
    int i = blockIdx.x * blockDim.x + threadIdx.x;
    if (i < n) {
        const float gi = gates[i];
        const float gf = gates[n + i];
        const float go = gates[2 * n + i];
        const float gc = gates[3 * n + i];
        const float cn = gf * cbuf[i] + gi * gc + (float)l_t[i] * (float)zh_t[i];
        const float h  = go * tanhf(cn);
        cbuf[i] = cn;
        hout[i] = h;
        hbf[i]  = (bf16)h;
    }
}

__global__ void cvt_f32_bf16_kernel(const float* __restrict__ src,
                                    bf16* __restrict__ dst, size_t n)
{
    size_t i = blockIdx.x * (size_t)blockDim.x + threadIdx.x;
    const size_t stride = (size_t)gridDim.x * blockDim.x;
    for (; i < n; i += stride) dst[i] = (bf16)src[i];
}

// src [K,N] f32 row-major  ->  dst [N,K] bf16 row-major
__global__ void transpose_cvt_kernel(const float* __restrict__ src,
                                     bf16* __restrict__ dst, int K, int N)
{
    const size_t n = (size_t)K * N;
    const size_t stride = (size_t)gridDim.x * blockDim.x;
    for (size_t idx = blockIdx.x * (size_t)blockDim.x + threadIdx.x; idx < n;
         idx += stride) {
        const int k = (int)(idx / N);
        const int c = (int)(idx % N);
        dst[(size_t)c * K + k] = (bf16)src[idx];
    }
}

// ---------------------------------------------------------------------------
// Host launcher
// ---------------------------------------------------------------------------
extern "C" void kernel_launch(void* const* d_in, const int* in_sizes, int n_in,
                              void* d_out, int out_size, void* d_ws, size_t ws_size,
                              hipStream_t stream)
{
    (void)in_sizes; (void)n_in; (void)out_size; (void)ws_size;

    const int T = 128, Bb = 256, Hh = 1024, Ee = 1024, Zz = 512;
    const size_t TB = (size_t)T * Bb;      // 32768
    const size_t BH = (size_t)Bb * Hh;     // 262144

    const float* input      = (const float*)d_in[0];
    const float* zin        = (const float*)d_in[1];
    const float* hidden     = (const float*)d_in[2];
    const float* cell_state = (const float*)d_in[3];
    const float* w_ih = (const float*)d_in[4];
    const float* w_oh = (const float*)d_in[5];
    const float* w_fh = (const float*)d_in[6];
    const float* w_ch = (const float*)d_in[7];
    const float* w_ix = (const float*)d_in[8];
    const float* w_ox = (const float*)d_in[9];
    const float* w_fx = (const float*)d_in[10];
    const float* w_cx = (const float*)d_in[11];
    const float* w_lx = (const float*)d_in[12];
    const float* w_zx = (const float*)d_in[13];
    const float* b_i  = (const float*)d_in[14];
    const float* b_f  = (const float*)d_in[15];
    const float* b_o  = (const float*)d_in[16];
    const float* b_l  = (const float*)d_in[17];
    const float* b_z  = (const float*)d_in[18];
    const float* b_c  = (const float*)d_in[19];

    // ---- workspace carve-up (256B aligned) ----
    char* wp = (char*)d_ws;
    auto carve = [&](size_t bytes) -> void* {
        void* p = (void*)wp;
        wp += (bytes + 255) & ~((size_t)255);
        return p;
    };
    bf16* Xin = (bf16*)carve(TB * Ee * sizeof(bf16));
    bf16* Zin = (bf16*)carve(TB * Zz * sizeof(bf16));
    bf16* Wxt[4]; for (int g = 0; g < 4; ++g) Wxt[g] = (bf16*)carve((size_t)Hh * Ee * sizeof(bf16));
    bf16* Wzt[2]; for (int g = 0; g < 2; ++g) Wzt[g] = (bf16*)carve((size_t)Hh * Zz * sizeof(bf16));
    bf16* Whc[4]; for (int g = 0; g < 4; ++g) Whc[g] = (bf16*)carve((size_t)Hh * Hh * sizeof(bf16));
    bf16* Xg[4];  for (int g = 0; g < 4; ++g) Xg[g]  = (bf16*)carve(TB * Hh * sizeof(bf16));
    bf16* Lb   = (bf16*)carve(TB * Hh * sizeof(bf16));
    bf16* Zhb  = (bf16*)carve(TB * Hh * sizeof(bf16));
    bf16* Hbf  = (bf16*)carve(BH * sizeof(bf16));
    float* Cbuf  = (float*)carve(BH * sizeof(float));
    float* Gates = (float*)carve(4 * BH * sizeof(float));

    // ---- phase 0: conversions / transposes to bf16 ----
    cvt_f32_bf16_kernel<<<4096, 256, 0, stream>>>(input, Xin, TB * Ee);
    cvt_f32_bf16_kernel<<<2048, 256, 0, stream>>>(zin, Zin, TB * Zz);
    transpose_cvt_kernel<<<1024, 256, 0, stream>>>(w_ix, Wxt[0], Ee, Hh);
    transpose_cvt_kernel<<<1024, 256, 0, stream>>>(w_fx, Wxt[1], Ee, Hh);
    transpose_cvt_kernel<<<1024, 256, 0, stream>>>(w_ox, Wxt[2], Ee, Hh);
    transpose_cvt_kernel<<<1024, 256, 0, stream>>>(w_cx, Wxt[3], Ee, Hh);
    transpose_cvt_kernel<<<512, 256, 0, stream>>>(w_lx, Wzt[0], Zz, Hh);
    transpose_cvt_kernel<<<512, 256, 0, stream>>>(w_zx, Wzt[1], Zz, Hh);
    // recurrent weights: h @ W^T needs Bt[n][k] = W[n][k] -> plain convert
    cvt_f32_bf16_kernel<<<1024, 256, 0, stream>>>(w_ih, Whc[0], (size_t)Hh * Hh);
    cvt_f32_bf16_kernel<<<1024, 256, 0, stream>>>(w_fh, Whc[1], (size_t)Hh * Hh);
    cvt_f32_bf16_kernel<<<1024, 256, 0, stream>>>(w_oh, Whc[2], (size_t)Hh * Hh);
    cvt_f32_bf16_kernel<<<1024, 256, 0, stream>>>(w_ch, Whc[3], (size_t)Hh * Hh);
    cvt_f32_bf16_kernel<<<256, 256, 0, stream>>>(hidden, Hbf, BH);
    hipMemcpyAsync(Cbuf, cell_state, BH * sizeof(float),
                   hipMemcpyDeviceToDevice, stream);

    // ---- phase 1: big input-projection GEMMs (bias fused; l/zhat activated) ----
    {
        dim3 grid(Hh / BN, (unsigned)(TB / BM));  // 8 x 256
        wmma_gemm_kernel<0><<<grid, 256, 0, stream>>>(Xin, Wxt[0], b_i, Xg[0], Hh, Ee);
        wmma_gemm_kernel<0><<<grid, 256, 0, stream>>>(Xin, Wxt[1], b_f, Xg[1], Hh, Ee);
        wmma_gemm_kernel<0><<<grid, 256, 0, stream>>>(Xin, Wxt[2], b_o, Xg[2], Hh, Ee);
        wmma_gemm_kernel<0><<<grid, 256, 0, stream>>>(Xin, Wxt[3], b_c, Xg[3], Hh, Ee);
        wmma_gemm_kernel<1><<<grid, 256, 0, stream>>>(Zin, Wzt[0], b_l, Lb,  Hh, Zz);
        wmma_gemm_kernel<2><<<grid, 256, 0, stream>>>(Zin, Wzt[1], b_z, Zhb, Hh, Zz);
    }

    // ---- phase 2: sequential scan over T ----
    dim3 rgrid(Hh / BN, Bb / BM, 4);  // 8 x 2 x 4 gates
    const int pwBlocks = (int)((BH + 255) / 256);
    float* hout = (float*)d_out;
    for (int t = 0; t < T; ++t) {
        const size_t toff = (size_t)t * BH;
        recurrent_gemm_kernel<<<rgrid, 256, 0, stream>>>(
            Hbf, Whc[0], Whc[1], Whc[2], Whc[3],
            Xg[0] + toff, Xg[1] + toff, Xg[2] + toff, Xg[3] + toff,
            Gates, Bb, Hh);
        pointwise_step_kernel<<<pwBlocks, 256, 0, stream>>>(
            Gates, Lb + toff, Zhb + toff, Cbuf, hout + toff, Hbf, (int)BH);
    }
}
